// Llama4TextDecoderLayer_1614907703553
// MI455X (gfx1250) — matrix operand order repaired and checked by reading, not compiled, not run on hardware
//
#include <hip/hip_runtime.h>
#include <math.h>

// ---------------------------------------------------------------------------
// Llama4 decoder layer for MI455X (gfx1250, wave32, WMMA bf16).
// fp32 weights are converted to bf16 with v_cvt_pk_bf16_f32 (inline asm)
// while staging into LDS; V_WMMA_F32_16X16X32_BF16 with f32 accumulate.
// Staging is software-pipelined; global accesses use 32-bit offsets so the
// backend can use saddr+voffset addressing.
// ---------------------------------------------------------------------------

#define T_TOK 2048
#define DIM   2048
#define S_SEQ 2048
#define NHQ   16
#define NKV   4
#define HDIM  128
#define NEXP  8
#define FDIM  2048
#define GUF   (2 * FDIM)

typedef __attribute__((ext_vector_type(16))) __bf16 v16bf;
typedef __attribute__((ext_vector_type(8)))  float  v8f;

union Frag16 { v16bf v; unsigned int u[8]; };

__device__ __forceinline__ float bf2f(unsigned short h) {
  return __uint_as_float(((unsigned int)h) << 16);
}
// two f32 -> two bf16 packed in one dword, hardware RNE conversion
__device__ __forceinline__ unsigned int pack2bf(float lo, float hi) {
  unsigned int r;
  asm volatile("v_cvt_pk_bf16_f32 %0, %1, %2" : "=v"(r) : "v"(lo), "v"(hi));
  return r;
}
__device__ __forceinline__ unsigned short f2bf(float f) {
  return (unsigned short)pack2bf(f, 0.0f);
}

enum { EPI_F32 = 0, EPI_BF16 = 1, EPI_RES_F32 = 2 };

// ---------------------------------------------------------------------------
// Tiled WMMA GEMM.  A: bf16 MxK row-major.  B: fp32.  C: f32/bf16.
// BNT==0: B is KxN row-major (NN).  BNT==1: B is NxK row-major (A * B^T).
// GUARD: bounds-check rows against M (grouped-MoE only).  N must be a
// multiple of 64 and K a multiple of 32 in all cases (true for this layer).
// Grouped-MoE mode: grpCnt/grpOff per blockIdx.z expert, B += e*strideB.
// causalSkip: drop tiles fully above the diagonal (score GEMM).
// causalK:    clamp K loop to mbase+64 (P*V GEMM; probs are 0 beyond).
// ---------------------------------------------------------------------------
template<int EPI, int BNT, bool GUARD>
__global__ void __launch_bounds__(128)
k_gemm(const unsigned short* __restrict__ A, int lda,
       const float* __restrict__ B, int ldb,
       void* __restrict__ C, int ldc,
       const float* __restrict__ Res,
       int M, int N, int K,
       int causalSkip, int causalK,
       const int* __restrict__ grpCnt, const int* __restrict__ grpOff,
       size_t strideB)
{
  if (GUARD) {
    int e = blockIdx.z;
    int cnt = grpCnt[e];
    if ((int)blockIdx.y * 64 >= cnt) return;
    M = cnt;
    size_t ro = (size_t)grpOff[e];
    A += ro * (size_t)lda;
    C = (void*)((float*)C + ro * (size_t)ldc);
    B += (size_t)e * strideB;
  }
  int mbase = blockIdx.y * 64;
  int nbase = blockIdx.x * 64;
  if (mbase >= M) return;
  if (causalSkip && nbase > mbase + 63) return;
  int Keff = K;
  if (causalK) { int lim = mbase + 64; if (lim < Keff) Keff = lim; }

  __shared__ unsigned int As[64][20];   // 64 rows x 32 bf16 (+pad), K-contig
  __shared__ unsigned int Bs[64][20];   // 64 N-rows x 32 bf16 (+pad), K-contig

  int tid  = threadIdx.x;
  int lane = tid & 31;
  int wave = tid >> 5;
  int half16 = lane >> 4;
  int l15    = lane & 15;

  v8f acc[4];
#pragma unroll
  for (int nt = 0; nt < 4; ++nt)
#pragma unroll
    for (int e = 0; e < 8; ++e) acc[nt][e] = 0.0f;

  // ---- staging registers (software pipeline) ----
  uint4  aReg[2];
  float4 bReg[4];

  auto loadTiles = [&](int kb) {
    // A: 64x32 bf16 = 256 x 16B; 2 uint4 per thread, coalesced along K
#pragma unroll
    for (int i = 0; i < 2; ++i) {
      int idx = tid + i * 128;
      int r = idx >> 2, q = idx & 3;               // q: 16B chunk within row
      int gr = mbase + r;
      unsigned aoff = (unsigned)gr * (unsigned)lda + (unsigned)(kb + q * 8);
      if (!GUARD || gr < M)
        aReg[i] = *(const uint4*)(A + aoff);
      else
        aReg[i] = uint4{0u, 0u, 0u, 0u};
    }
    if (BNT == 0) {
      // B: 32(K) x 64(N) f32; thread takes a k-row PAIR x 4 columns
#pragma unroll
      for (int i = 0; i < 2; ++i) {
        int idx = tid + i * 128;
        int kr2 = idx >> 4;                        // k pair index 0..15
        int n4  = (idx & 15) << 2;
        unsigned boff = (unsigned)(kb + kr2 * 2) * (unsigned)ldb +
                        (unsigned)(nbase + n4);
        bReg[2 * i]     = *(const float4*)(B + boff);
        bReg[2 * i + 1] = *(const float4*)(B + boff + (unsigned)ldb);
      }
    } else {
      // B: 64(N) x 32(K) f32; float4 along contiguous K
#pragma unroll
      for (int i = 0; i < 4; ++i) {
        int idx = tid + i * 128;
        int n = idx >> 3;
        int q = (idx & 7) << 2;
        unsigned boff = (unsigned)(nbase + n) * (unsigned)ldb +
                        (unsigned)(kb + q);
        bReg[i] = *(const float4*)(B + boff);
      }
    }
  };

  auto storeTiles = [&]() {
#pragma unroll
    for (int i = 0; i < 2; ++i) {
      int idx = tid + i * 128;
      int r = idx >> 2, q = idx & 3;
      *(uint4*)&As[r][q << 2] = aReg[i];
    }
    if (BNT == 0) {
      // transpose to Bs[n][k]: dword kr2 holds bf16 pair (k=2*kr2, 2*kr2+1)
#pragma unroll
      for (int i = 0; i < 2; ++i) {
        int idx = tid + i * 128;
        int kr2 = idx >> 4;
        int n4  = (idx & 15) << 2;
        float4 b0 = bReg[2 * i];                   // k = 2*kr2
        float4 b1 = bReg[2 * i + 1];               // k = 2*kr2+1
        Bs[n4 + 0][kr2] = pack2bf(b0.x, b1.x);
        Bs[n4 + 1][kr2] = pack2bf(b0.y, b1.y);
        Bs[n4 + 2][kr2] = pack2bf(b0.z, b1.z);
        Bs[n4 + 3][kr2] = pack2bf(b0.w, b1.w);
      }
    } else {
#pragma unroll
      for (int i = 0; i < 4; ++i) {
        int idx = tid + i * 128;
        int n = idx >> 3;
        int q = (idx & 7) << 2;                    // k offset (halves)
        uint2 p;
        p.x = pack2bf(bReg[i].x, bReg[i].y);
        p.y = pack2bf(bReg[i].z, bReg[i].w);
        *(uint2*)&Bs[n][q >> 1] = p;
      }
    }
  };

  loadTiles(0);
  for (int kb = 0; kb < Keff; kb += 32) {
    storeTiles();
    __syncthreads();
    if (kb + 32 < Keff) loadTiles(kb + 32);        // prefetch next tile

    // A fragment: 16x32 bf16, ISA layout (row = lane&15; K split lane halves)
    Frag16 fa;
#pragma unroll
    for (int v = 0; v < 8; ++v) {
      int k = ((v & 4) << 2) | (half16 << 3) | ((v & 3) << 1);
      fa.u[v] = As[wave * 16 + l15][k >> 1];
    }
#pragma unroll
    for (int nt = 0; nt < 4; ++nt) {
      // B fragment: 32x16 bf16 (col = lane&15; K split lane halves, same as A)
      Frag16 fb;
#pragma unroll
      for (int v = 0; v < 8; ++v) {
        int k = ((v & 4) << 2) | (half16 << 3) | ((v & 3) << 1);
        fb.u[v] = Bs[nt * 16 + l15][k >> 1];
      }
      acc[nt] = __builtin_amdgcn_wmma_f32_16x16x32_bf16(
          false, fa.v, false, fb.v, (short)0, acc[nt], false, false);
    }
    __syncthreads();
  }

  // epilogue: C layout -> lanes 0-15: M=r, N=lane; lanes 16-31: M=r+8
#pragma unroll
  for (int nt = 0; nt < 4; ++nt) {
#pragma unroll
    for (int r = 0; r < 8; ++r) {
      int row = mbase + wave * 16 + r + (half16 ? 8 : 0);
      int col = nbase + nt * 16 + l15;
      if (!GUARD || row < M) {
        float vv = acc[nt][r];
        unsigned coff = (unsigned)row * (unsigned)ldc + (unsigned)col;
        if (EPI == EPI_F32)
          ((float*)C)[coff] = vv;
        else if (EPI == EPI_BF16)
          ((unsigned short*)C)[coff] = f2bf(vv);
        else
          ((float*)C)[coff] = Res[coff] + vv;
      }
    }
  }
}

// ---------------------------------------------------------------------------
// RMSNorm (f32 in, bf16 out), pair-packed stores
// ---------------------------------------------------------------------------
__global__ void k_rmsnorm(const float* __restrict__ x,
                          const float* __restrict__ w,
                          unsigned short* __restrict__ out)
{
  int t = blockIdx.x, tid = threadIdx.x;
  const float* row = x + (size_t)t * DIM;
  float ss = 0.0f;
  for (int d = tid * 2; d < DIM; d += 512) {
    float2 v = *(const float2*)&row[d];
    ss += v.x * v.x + v.y * v.y;
  }
  __shared__ float red[256];
  red[tid] = ss; __syncthreads();
  for (int st = 128; st > 0; st >>= 1) {
    if (tid < st) red[tid] += red[tid + st];
    __syncthreads();
  }
  float scale = rsqrtf(red[0] / (float)DIM + 1e-5f);
  for (int d = tid * 2; d < DIM; d += 512) {
    float2 v = *(const float2*)&row[d];
    float2 ww = *(const float2*)&w[d];
    *(unsigned int*)&out[(size_t)t * DIM + d] =
        pack2bf(v.x * scale * ww.x, v.y * scale * ww.y);
  }
}

// ---------------------------------------------------------------------------
// RoPE + per-head L2 norm.  grid(heads, S), block 64 (one rotation pair each)
// ---------------------------------------------------------------------------
template<bool OUT_BF16>
__global__ void k_rope_l2(const float* __restrict__ x, int ldx,
                          const float* __restrict__ cosp,
                          const float* __restrict__ sinp,
                          void* __restrict__ out, int ldo)
{
  int h = blockIdx.x, s = blockIdx.y, i = threadIdx.x;  // i in [0,64)
  const float* base = x + (size_t)s * ldx + h * HDIM;
  float a = base[2 * i], b = base[2 * i + 1];
  float c  = cosp[(size_t)s * 64 + i];
  float sn = sinp[(size_t)s * 64 + i];
  float na = a * c - b * sn;
  float nb = a * sn + b * c;
  __shared__ float red[64];
  red[i] = na * na + nb * nb;
  __syncthreads();
  for (int st = 32; st > 0; st >>= 1) {
    if (i < st) red[i] += red[i + st];
    __syncthreads();
  }
  float scale = rsqrtf(red[0] / (float)HDIM + 1e-6f);
  if (OUT_BF16) {
    unsigned short* o = (unsigned short*)out + (size_t)s * ldo + h * HDIM;
    *(unsigned int*)&o[2 * i] = pack2bf(na * scale, nb * scale);
  } else {
    float* o = (float*)out + (size_t)s * ldo + h * HDIM;
    o[2 * i]     = na * scale;
    o[2 * i + 1] = nb * scale;
  }
}

// ---------------------------------------------------------------------------
// Causal softmax: one row per block; probs out in bf16 (zeros past diagonal)
// ---------------------------------------------------------------------------
__global__ void k_softmax_causal(const float* __restrict__ sc,
                                 unsigned short* __restrict__ p, float scale)
{
  int r = blockIdx.x, tid = threadIdx.x;
  int nvalid = r + 1;
  const float* row = sc + (size_t)r * S_SEQ;
  __shared__ float red[256];
  float mx = -3.0e38f;
  for (int c = tid; c < nvalid; c += 256) mx = fmaxf(mx, row[c] * scale);
  red[tid] = mx; __syncthreads();
  for (int st = 128; st > 0; st >>= 1) {
    if (tid < st) red[tid] = fmaxf(red[tid], red[tid + st]);
    __syncthreads();
  }
  mx = red[0]; __syncthreads();
  float sum = 0.0f;
  for (int c = tid; c < nvalid; c += 256) sum += __expf(row[c] * scale - mx);
  red[tid] = sum; __syncthreads();
  for (int st = 128; st > 0; st >>= 1) {
    if (tid < st) red[tid] += red[tid + st];
    __syncthreads();
  }
  float inv = 1.0f / red[0];
  for (int c = tid * 2; c < S_SEQ; c += 512) {
    float v0 = (c     < nvalid) ? __expf(row[c]     * scale - mx) * inv : 0.0f;
    float v1 = (c + 1 < nvalid) ? __expf(row[c + 1] * scale - mx) * inv : 0.0f;
    *(unsigned int*)&p[(size_t)r * S_SEQ + c] = pack2bf(v0, v1);
  }
}

// ---------------------------------------------------------------------------
// Router: logits = t2 @ router_w (D x 8), top-1 + sigmoid gate
// ---------------------------------------------------------------------------
__global__ void k_router(const unsigned short* __restrict__ t2,
                         const float* __restrict__ rw,
                         int* __restrict__ topi, float* __restrict__ gate)
{
  int t = blockIdx.x, tid = threadIdx.x;
  float acc[NEXP];
#pragma unroll
  for (int e = 0; e < NEXP; ++e) acc[e] = 0.0f;
  for (int d = tid; d < DIM; d += 256) {
    float x = bf2f(t2[(size_t)t * DIM + d]);
    const float* w = rw + (size_t)d * NEXP;
#pragma unroll
    for (int e = 0; e < NEXP; ++e) acc[e] += x * w[e];
  }
  __shared__ float red[NEXP][256];
#pragma unroll
  for (int e = 0; e < NEXP; ++e) red[e][tid] = acc[e];
  __syncthreads();
  for (int st = 128; st > 0; st >>= 1) {
    if (tid < st)
#pragma unroll
      for (int e = 0; e < NEXP; ++e) red[e][tid] += red[e][tid + st];
    __syncthreads();
  }
  if (tid == 0) {
    float best = red[0][0]; int bi = 0;
    for (int e = 1; e < NEXP; ++e) {
      float v = red[e][0];
      if (v > best) { best = v; bi = e; }
    }
    topi[t] = bi;
    gate[t] = 1.0f / (1.0f + __expf(-best));
  }
}

__global__ void k_zero_counts(int* counts) {
  if (threadIdx.x < NEXP) counts[threadIdx.x] = 0;
}
__global__ void k_count(const int* __restrict__ topi, int* __restrict__ counts) {
  int t = blockIdx.x * 256 + threadIdx.x;
  if (t < T_TOK) atomicAdd(&counts[topi[t]], 1);
}
__global__ void k_scan(const int* __restrict__ counts, int* __restrict__ offs,
                       int* __restrict__ fill, int* __restrict__ total) {
  int s = 0;
  for (int e = 0; e < NEXP; ++e) { offs[e] = s; fill[e] = s; s += counts[e]; }
  *total = s;
}
__global__ void k_assign(const int* __restrict__ topi, int* __restrict__ fill,
                         int* __restrict__ slot, int* __restrict__ perm) {
  int t = blockIdx.x * 256 + threadIdx.x;
  if (t < T_TOK) {
    int s = atomicAdd(&fill[topi[t]], 1);
    slot[t] = s;
    perm[s] = t;
  }
}
__global__ void k_gather(const unsigned short* __restrict__ t2,
                         const float* __restrict__ gate,
                         const int* __restrict__ slot,
                         unsigned short* __restrict__ Xg) {
  int t = blockIdx.y;
  int d = (blockIdx.x * 256 + threadIdx.x) * 2;
  int sl = slot[t];
  float g = gate[t];
  unsigned int tw = *(const unsigned int*)&t2[(size_t)t * DIM + d];
  *(unsigned int*)&Xg[(size_t)sl * DIM + d] =
      pack2bf(bf2f((unsigned short)tw) * g,
              bf2f((unsigned short)(tw >> 16)) * g);
}
__global__ void k_silu_gu(const float* __restrict__ GU,
                          unsigned short* __restrict__ Hg,
                          const int* __restrict__ total) {
  int sl = blockIdx.y;
  if (sl >= *total) return;
  int f = (blockIdx.x * 256 + threadIdx.x) * 2;
  float2 g2 = *(const float2*)&GU[(size_t)sl * GUF + f];
  float2 u2 = *(const float2*)&GU[(size_t)sl * GUF + FDIM + f];
  *(unsigned int*)&Hg[(size_t)sl * FDIM + f] =
      pack2bf(u2.x * g2.x / (1.0f + __expf(-g2.x)),
              u2.y * g2.y / (1.0f + __expf(-g2.y)));
}
__global__ void k_scatter(const int* __restrict__ perm,
                          const int* __restrict__ total,
                          const float* __restrict__ Yg,
                          float* __restrict__ moe) {
  int sl = blockIdx.y;
  if (sl >= *total) return;
  int d = (blockIdx.x * 256 + threadIdx.x) * 4;
  int t = perm[sl];
  float4 a = *(const float4*)&moe[(size_t)t * DIM + d];
  float4 b = *(const float4*)&Yg[(size_t)sl * DIM + d];
  a.x += b.x; a.y += b.y; a.z += b.z; a.w += b.w;
  *(float4*)&moe[(size_t)t * DIM + d] = a;
}
__global__ void k_silu_mul(const float* __restrict__ g,
                           const float* __restrict__ u,
                           unsigned short* __restrict__ o) {
  int t = blockIdx.y;
  int f = (blockIdx.x * 256 + threadIdx.x) * 2;
  float2 g2 = *(const float2*)&g[(size_t)t * FDIM + f];
  float2 u2 = *(const float2*)&u[(size_t)t * FDIM + f];
  *(unsigned int*)&o[(size_t)t * FDIM + f] =
      pack2bf(g2.x / (1.0f + __expf(-g2.x)) * u2.x,
              g2.y / (1.0f + __expf(-g2.y)) * u2.y);
}
__global__ void k_add_out(const float* __restrict__ a,
                          const float* __restrict__ b,
                          float* __restrict__ o) {
  int t = blockIdx.y;
  int d = (blockIdx.x * 256 + threadIdx.x) * 4;
  float4 x = *(const float4*)&a[(size_t)t * DIM + d];
  float4 y = *(const float4*)&b[(size_t)t * DIM + d];
  x.x += y.x; x.y += y.y; x.z += y.z; x.w += y.w;
  *(float4*)&o[(size_t)t * DIM + d] = x;
}

// ---------------------------------------------------------------------------
// Workspace layout (bytes, 256-aligned)
// ---------------------------------------------------------------------------
static constexpr size_t alup(size_t x) { return (x + 255) & ~(size_t)255; }
static constexpr size_t SZ_T_BF = (size_t)T_TOK * DIM * 2;    // 8 MiB
static constexpr size_t SZ_TD_F = (size_t)T_TOK * DIM * 4;    // 16 MiB
static constexpr size_t SZ_KV_F = (size_t)T_TOK * (NKV * HDIM) * 4;  // 4 MiB
static constexpr size_t SZ_GU_F = (size_t)T_TOK * GUF * 4;    // 32 MiB

static constexpr size_t O_T1  = 0;                          // t1 bf16 -> t2 bf16
static constexpr size_t O_QF  = O_T1  + alup(SZ_T_BF);      // q f32 -> shared gate f32
static constexpr size_t O_KF  = O_QF  + alup(SZ_TD_F);      // k f32
static constexpr size_t O_VF  = O_KF  + alup(SZ_KV_F);      // v f32
static constexpr size_t O_QBF = O_VF  + alup(SZ_KV_F);      // q bf16 -> Xg bf16
static constexpr size_t O_SC  = O_QBF + alup(SZ_T_BF);      // scores f32 -> shared up f32
static constexpr size_t O_PR  = O_SC  + alup(SZ_TD_F);      // probs bf16 -> hgu bf16
static constexpr size_t O_ATT = O_PR  + alup(SZ_T_BF);      // attn bf16 -> Hg bf16
static constexpr size_t O_HS2 = O_ATT + alup(SZ_T_BF);      // hs2 f32 (residual2)
static constexpr size_t O_MOE = O_HS2 + alup(SZ_TD_F);      // moe f32
static constexpr size_t O_GU  = O_MOE + alup(SZ_TD_F);      // routed gate_up out f32
static constexpr size_t O_YG  = O_GU  + alup(SZ_GU_F);      // routed down out f32
static constexpr size_t O_INT = O_YG  + alup(SZ_TD_F);      // router scratch

extern "C" void kernel_launch(void* const* d_in, const int* in_sizes, int n_in,
                              void* d_out, int out_size, void* d_ws, size_t ws_size,
                              hipStream_t stream)
{
  (void)in_sizes; (void)n_in; (void)out_size; (void)ws_size;
  const float* hidden = (const float*)d_in[0];
  const float* fcos   = (const float*)d_in[1];
  const float* fsin   = (const float*)d_in[2];
  // d_in[3] attention_mask: causal, applied analytically in softmax
  // d_in[4] cache_position: unused
  const float* wq  = (const float*)d_in[5];
  const float* wk  = (const float*)d_in[6];
  const float* wv  = (const float*)d_in[7];
  const float* wo  = (const float*)d_in[8];
  const float* ln1 = (const float*)d_in[9];
  const float* ln2 = (const float*)d_in[10];
  const float* rw  = (const float*)d_in[11];
  const float* gup = (const float*)d_in[12];
  const float* dwn = (const float*)d_in[13];
  const float* sg  = (const float*)d_in[14];
  const float* su  = (const float*)d_in[15];
  const float* sd  = (const float*)d_in[16];
  float* out = (float*)d_out;

  char* ws = (char*)d_ws;
  unsigned short* t1   = (unsigned short*)(ws + O_T1);   // also t2
  float*          qf   = (float*)(ws + O_QF);            // also shared-gate
  float*          kf   = (float*)(ws + O_KF);
  float*          vf   = (float*)(ws + O_VF);
  unsigned short* qbf  = (unsigned short*)(ws + O_QBF);  // also Xg
  float*          scb  = (float*)(ws + O_SC);            // also shared-up
  unsigned short* prb  = (unsigned short*)(ws + O_PR);   // also hgu
  unsigned short* attb = (unsigned short*)(ws + O_ATT);  // also Hg
  float*          hs2  = (float*)(ws + O_HS2);
  float*          moe  = (float*)(ws + O_MOE);
  float*          GUo  = (float*)(ws + O_GU);
  float*          Yg   = (float*)(ws + O_YG);
  int* counts = (int*)(ws + O_INT);
  int* offs   = counts + 8;
  int* fill   = offs + 8;
  int* total  = fill + 8;
  int* topi   = total + 8;                 // 2048 ints
  int* slot   = topi + T_TOK;
  int* perm   = slot + T_TOK;
  float* gatef = (float*)(perm + T_TOK);

  const float scaling = 0.08838834764831843f;  // 1/sqrt(HD)
  dim3 b128(128), b256(256), b64(64);

  // ---- 1. RMSNorm 1 -> t1 (bf16)
  k_rmsnorm<<<T_TOK, b256, 0, stream>>>(hidden, ln1, t1);

  // ---- 2. QKV projections
  k_gemm<EPI_F32, 0, false><<<dim3(DIM / 64, T_TOK / 64), b128, 0, stream>>>(
      t1, DIM, wq, DIM, qf, DIM, nullptr, T_TOK, DIM, DIM, 0, 0, nullptr, nullptr, 0);
  k_gemm<EPI_F32, 0, false><<<dim3((NKV * HDIM) / 64, T_TOK / 64), b128, 0, stream>>>(
      t1, DIM, wk, NKV * HDIM, kf, NKV * HDIM, nullptr, T_TOK, NKV * HDIM, DIM, 0, 0, nullptr, nullptr, 0);
  k_gemm<EPI_F32, 0, false><<<dim3((NKV * HDIM) / 64, T_TOK / 64), b128, 0, stream>>>(
      t1, DIM, wv, NKV * HDIM, vf, NKV * HDIM, nullptr, T_TOK, NKV * HDIM, DIM, 0, 0, nullptr, nullptr, 0);

  // ---- 3. RoPE + L2 norm (q -> bf16, k -> f32 in place)
  k_rope_l2<true><<<dim3(NHQ, S_SEQ), b64, 0, stream>>>(qf, DIM, fcos, fsin, qbf, DIM);
  k_rope_l2<false><<<dim3(NKV, S_SEQ), b64, 0, stream>>>(kf, NKV * HDIM, fcos, fsin, kf, NKV * HDIM);

  // ---- 4. Attention, one head at a time (GQA: kv head = h/4)
  for (int h = 0; h < NHQ; ++h) {
    int kvh = h / (NHQ / NKV);
    // scores = Qh * Kh^T (causal tile skip)
    k_gemm<EPI_F32, 1, false><<<dim3(S_SEQ / 64, S_SEQ / 64), b128, 0, stream>>>(
        qbf + h * HDIM, DIM, kf + kvh * HDIM, NKV * HDIM, scb, S_SEQ,
        nullptr, S_SEQ, S_SEQ, HDIM, 1, 0, nullptr, nullptr, 0);
    // softmax(scale * scores) with causal mask -> probs (bf16)
    k_softmax_causal<<<S_SEQ, b256, 0, stream>>>(scb, prb, scaling);
    // attn = probs * Vh (K loop clamped past diagonal)
    k_gemm<EPI_BF16, 0, false><<<dim3(HDIM / 64, S_SEQ / 64), b128, 0, stream>>>(
        prb, S_SEQ, vf + kvh * HDIM, NKV * HDIM, attb + h * HDIM, DIM,
        nullptr, S_SEQ, HDIM, S_SEQ, 0, 1, nullptr, nullptr, 0);
  }

  // ---- 5. Output projection + residual -> hs2
  k_gemm<EPI_RES_F32, 0, false><<<dim3(DIM / 64, T_TOK / 64), b128, 0, stream>>>(
      attb, DIM, wo, DIM, hs2, DIM, hidden, T_TOK, DIM, DIM, 0, 0, nullptr, nullptr, 0);

  // ---- 6. RMSNorm 2 -> t2 (bf16, reuses t1 region)
  k_rmsnorm<<<T_TOK, b256, 0, stream>>>(hs2, ln2, t1);

  // ---- 7. Router top-1 + grouped-token gather
  k_router<<<T_TOK, b256, 0, stream>>>(t1, rw, topi, gatef);
  k_zero_counts<<<1, 32, 0, stream>>>(counts);
  k_count<<<T_TOK / 256, b256, 0, stream>>>(topi, counts);
  k_scan<<<1, 1, 0, stream>>>(counts, offs, fill, total);
  k_assign<<<T_TOK / 256, b256, 0, stream>>>(topi, fill, slot, perm);
  k_gather<<<dim3(DIM / 512, T_TOK), b256, 0, stream>>>(t1, gatef, slot, qbf /*Xg*/);

  // ---- 8. Routed experts: grouped GEMMs (worst-case grid, device-count exit)
  k_gemm<EPI_F32, 0, true><<<dim3(GUF / 64, T_TOK / 64, NEXP), b128, 0, stream>>>(
      qbf /*Xg*/, DIM, gup, GUF, GUo, GUF, nullptr, T_TOK, GUF, DIM, 0, 0,
      counts, offs, (size_t)DIM * GUF);
  k_silu_gu<<<dim3(FDIM / 512, T_TOK), b256, 0, stream>>>(GUo, attb /*Hg*/, total);
  k_gemm<EPI_F32, 0, true><<<dim3(DIM / 64, T_TOK / 64, NEXP), b128, 0, stream>>>(
      attb /*Hg*/, FDIM, dwn, DIM, Yg, DIM, nullptr, T_TOK, DIM, FDIM, 0, 0,
      counts, offs, (size_t)FDIM * DIM);

  // ---- 9. Shared expert -> moe
  k_gemm<EPI_F32, 0, false><<<dim3(FDIM / 64, T_TOK / 64), b128, 0, stream>>>(
      t1, DIM, sg, FDIM, qf /*g*/, FDIM, nullptr, T_TOK, FDIM, DIM, 0, 0, nullptr, nullptr, 0);
  k_gemm<EPI_F32, 0, false><<<dim3(FDIM / 64, T_TOK / 64), b128, 0, stream>>>(
      t1, DIM, su, FDIM, scb /*u*/, FDIM, nullptr, T_TOK, FDIM, DIM, 0, 0, nullptr, nullptr, 0);
  k_silu_mul<<<dim3(FDIM / 512, T_TOK), b256, 0, stream>>>(qf, scb, prb /*hgu*/);
  k_gemm<EPI_F32, 0, false><<<dim3(DIM / 64, T_TOK / 64), b128, 0, stream>>>(
      prb /*hgu*/, FDIM, sd, DIM, moe, DIM, nullptr, T_TOK, DIM, FDIM, 0, 0, nullptr, nullptr, 0);

  // ---- 10. Scatter routed outputs into moe, final residual add
  k_scatter<<<dim3(DIM / 1024, T_TOK), b256, 0, stream>>>(perm, total, Yg, moe);
  k_add_out<<<dim3(DIM / 1024, T_TOK), b256, 0, stream>>>(hs2, moe, out);
}
